// AttentionBlock_84868553769003
// MI455X (gfx1250) — matrix operand order repaired and checked by reading, not compiled, not run on hardware
//
#include <hip/hip_runtime.h>

typedef _Float16 half_t;
typedef _Float16 v16h __attribute__((ext_vector_type(16)));
typedef _Float16 v8h  __attribute__((ext_vector_type(8)));
typedef float    v8f  __attribute__((ext_vector_type(8)));

#define BB 4
#define CC 64
#define NN 4096   // H*W
#define OC 192    // 3*C

// contiguous 16 halves -> B operand (per-lane)
__device__ __forceinline__ v16h ld16(const half_t* p) {
  v8h lo = *(const v8h*)(p);
  v8h hi = *(const v8h*)(p + 8);
  v16h r;
#pragma unroll
  for (int i = 0; i < 8; ++i) { r[i] = lo[i]; r[i + 8] = hi[i]; }
  return r;
}
// A operand: {p[0..7], p[16..23]}  (lane-half handled by caller via +8h offset)
__device__ __forceinline__ v16h ld_a(const half_t* p) {
  v8h lo = *(const v8h*)(p);
  v8h hi = *(const v8h*)(p + 16);
  v16h r;
#pragma unroll
  for (int i = 0; i < 8; ++i) { r[i] = lo[i]; r[i + 8] = hi[i]; }
  return r;
}
__device__ __forceinline__ v8f wmma_f16(v16h a, v16h b, v8f c) {
  return __builtin_amdgcn_wmma_f32_16x16x32_f16(false, a, false, b, (short)0, c,
                                                false, false);
}

// ---------------- Stage 1: fp32 -> f16 conversions (+ x transpose) ------------
__global__ void cvt_kernel(const float* __restrict__ x,
                           const float* __restrict__ wqkv,
                           const float* __restrict__ wproj,
                           half_t* __restrict__ xt,
                           half_t* __restrict__ wqh,
                           half_t* __restrict__ wph) {
  int idx = blockIdx.x * blockDim.x + threadIdx.x;
  if (idx < BB * CC * NN) {
    int b = idx / (CC * NN);
    int r = idx - b * CC * NN;
    int c = r / NN;
    int n = r - c * NN;
    xt[(size_t)b * NN * CC + (size_t)n * CC + c] = (half_t)x[idx];
  }
  if (idx < OC * CC) wqh[idx] = (half_t)wqkv[idx];
  if (idx < CC * CC) wph[idx] = (half_t)wproj[idx];
}

// ---------------- Stage 2: qkv = w_qkv @ x + b (WMMA) -------------------------
// q -> qn[b][n][c]   (A-friendly)
// k -> kt[b][m][c]   (B-friendly for scores)
// v -> vh[b][c][m]   (B-friendly for P·V)
__global__ void qkv_kernel(const half_t* __restrict__ xt,
                           const half_t* __restrict__ wqh,
                           const float* __restrict__ bqkv,
                           half_t* __restrict__ qn,
                           half_t* __restrict__ kt,
                           half_t* __restrict__ vh) {
  int lane = threadIdx.x & 31;
  int wv = threadIdx.x >> 5;
  int tile = blockIdx.x * 8 + wv;              // BB * 12 * (NN/16) tiles
  int b = tile / (12 * (NN / 16));
  int rem = tile - b * (12 * (NN / 16));
  int ot = rem / (NN / 16);
  int nt = rem - ot * (NN / 16);
  int col = lane & 15;
  int h = lane >> 4;

  const half_t* wrow = wqh + (size_t)(ot * 16 + col) * CC;
  v16h a0 = ld16(wrow + 16 * h);
  v16h a1 = ld16(wrow + 32 + 16 * h);
  const half_t* xrow = xt + (size_t)b * NN * CC + (size_t)(nt * 16 + col) * CC;
  v16h b0 = ld16(xrow + 16 * h);
  v16h b1 = ld16(xrow + 32 + 16 * h);
  v8f acc = {};
  acc = wmma_f16(a0, b0, acc);
  acc = wmma_f16(a1, b1, acc);
#pragma unroll
  for (int r = 0; r < 8; ++r) acc[r] += bqkv[ot * 16 + r + 8 * h];

  if (ot < 4) {
    v8h pk;
#pragma unroll
    for (int r = 0; r < 8; ++r) pk[r] = (half_t)acc[r];
    *(v8h*)(qn + (size_t)b * NN * CC + (size_t)(nt * 16 + col) * CC + ot * 16 +
            8 * h) = pk;
  } else if (ot < 8) {
    v8h pk;
#pragma unroll
    for (int r = 0; r < 8; ++r) pk[r] = (half_t)acc[r];
    *(v8h*)(kt + (size_t)b * NN * CC + (size_t)(nt * 16 + col) * CC +
            (ot - 4) * 16 + 8 * h) = pk;
  } else {
    int cb = (ot - 8) * 16;
#pragma unroll
    for (int r = 0; r < 8; ++r)
      vh[(size_t)b * CC * NN + (size_t)(cb + r + 8 * h) * NN + nt * 16 + col] =
          (half_t)acc[r];
  }
}

// ---------------- Stage 3: fused flash attention (WMMA + online softmax) ------
__global__ void attn_kernel(const half_t* __restrict__ qn,
                            const half_t* __restrict__ kt,
                            const half_t* __restrict__ vh,
                            half_t* __restrict__ aout) {
  __shared__ __align__(64) half_t plds[8][16][32];
  int lane = threadIdx.x & 31;
  int wv = threadIdx.x >> 5;
  int tile = blockIdx.x * 8 + wv;              // BB * (NN/16) tiles
  int b = tile / (NN / 16);
  int nt = tile - b * (NN / 16);
  int col = lane & 15;
  int h = lane >> 4;

  const half_t* qrow = qn + (size_t)b * NN * CC + (size_t)(nt * 16 + col) * CC;
  v16h aq0 = ld16(qrow + 16 * h);
  v16h aq1 = ld16(qrow + 32 + 16 * h);

  const half_t* ktb = kt + (size_t)b * NN * CC;
  const half_t* vb = vh + (size_t)b * CC * NN;

  float M[8], L[8];
#pragma unroll
  for (int r = 0; r < 8; ++r) { M[r] = -1e30f; L[r] = 0.f; }
  v8f O0 = {}, O1 = {}, O2 = {}, O3 = {};
  const float scale = 0.125f;  // 1/sqrt(C)

  for (int mb = 0; mb < NN; mb += 32) {
    const half_t* k0 = ktb + (size_t)(mb + col) * CC;
    const half_t* k1 = ktb + (size_t)(mb + 16 + col) * CC;
    __builtin_prefetch(ktb + (size_t)(((mb + 32) & (NN - 1)) + col) * CC, 0, 3);

    v8f s0 = {}, s1 = {};
    s0 = wmma_f16(aq0, ld16(k0 + 16 * h), s0);
    s0 = wmma_f16(aq1, ld16(k0 + 32 + 16 * h), s0);
    s1 = wmma_f16(aq0, ld16(k1 + 16 * h), s1);
    s1 = wmma_f16(aq1, ld16(k1 + 32 + 16 * h), s1);

#pragma unroll
    for (int r = 0; r < 8; ++r) {
      float x0 = s0[r] * scale, x1 = s1[r] * scale;
      float mx = fmaxf(x0, x1);
#pragma unroll
      for (int off = 1; off < 16; off <<= 1)
        mx = fmaxf(mx, __shfl_xor(mx, off, 16));
      float nm = fmaxf(M[r], mx);
      float corr = __expf(M[r] - nm);
      float p0 = __expf(x0 - nm);
      float p1 = __expf(x1 - nm);
      float rs = p0 + p1;
#pragma unroll
      for (int off = 1; off < 16; off <<= 1) rs += __shfl_xor(rs, off, 16);
      L[r] = L[r] * corr + rs;
      M[r] = nm;
      O0[r] *= corr; O1[r] *= corr; O2[r] *= corr; O3[r] *= corr;
      plds[wv][r + 8 * h][col] = (half_t)p0;
      plds[wv][r + 8 * h][16 + col] = (half_t)p1;
    }
    // re-layout P (C/D format) -> A-operand format through LDS
    v16h ap = ld16(&plds[wv][col][0] + 16 * h);
    const half_t* v0 = vb + (size_t)col * NN + mb + 16 * h;
    O0 = wmma_f16(ap, ld16(v0), O0);
    O1 = wmma_f16(ap, ld16(v0 + (size_t)16 * NN), O1);
    O2 = wmma_f16(ap, ld16(v0 + (size_t)32 * NN), O2);
    O3 = wmma_f16(ap, ld16(v0 + (size_t)48 * NN), O3);
  }

  half_t* arow = aout + (size_t)b * NN * CC;
#pragma unroll
  for (int r = 0; r < 8; ++r) {
    float inv = 1.0f / L[r];
    size_t n = (size_t)(nt * 16 + r + 8 * h);
    arow[n * CC + col] = (half_t)(O0[r] * inv);
    arow[n * CC + 16 + col] = (half_t)(O1[r] * inv);
    arow[n * CC + 32 + col] = (half_t)(O2[r] * inv);
    arow[n * CC + 48 + col] = (half_t)(O3[r] * inv);
  }
}

// ---------------- Stage 4: proj + bias + residual (WMMA) ----------------------
__global__ void proj_kernel(const half_t* __restrict__ aout,
                            const half_t* __restrict__ wph,
                            const float* __restrict__ bproj,
                            const float* __restrict__ x,
                            float* __restrict__ out) {
  int lane = threadIdx.x & 31;
  int wv = threadIdx.x >> 5;
  int tile = blockIdx.x * 8 + wv;              // BB * 4 * (NN/16) tiles
  int b = tile / (4 * (NN / 16));
  int rem = tile - b * (4 * (NN / 16));
  int ot = rem / (NN / 16);
  int nt = rem - ot * (NN / 16);
  int col = lane & 15;
  int h = lane >> 4;

  const half_t* wrow = wph + (size_t)(ot * 16 + col) * CC;
  v16h a0 = ld16(wrow + 16 * h);
  v16h a1 = ld16(wrow + 32 + 16 * h);
  const half_t* arow = aout + (size_t)b * NN * CC + (size_t)(nt * 16 + col) * CC;
  v16h b0 = ld16(arow + 16 * h);
  v16h b1 = ld16(arow + 32 + 16 * h);
  v8f acc = {};
  acc = wmma_f16(a0, b0, acc);
  acc = wmma_f16(a1, b1, acc);
#pragma unroll
  for (int r = 0; r < 8; ++r) {
    int o = ot * 16 + r + 8 * h;
    size_t idx = (size_t)b * CC * NN + (size_t)o * NN + nt * 16 + col;
    out[idx] = acc[r] + bproj[o] + x[idx];
  }
}

// ---------------- launch ------------------------------------------------------
extern "C" void kernel_launch(void* const* d_in, const int* in_sizes, int n_in,
                              void* d_out, int out_size, void* d_ws,
                              size_t ws_size, hipStream_t stream) {
  const float* x = (const float*)d_in[0];
  const float* wqkv = (const float*)d_in[1];
  const float* bqkv = (const float*)d_in[2];
  const float* wproj = (const float*)d_in[3];
  const float* bproj = (const float*)d_in[4];
  float* out = (float*)d_out;

  char* ws = (char*)d_ws;
  const size_t SZ = (size_t)BB * NN * CC * sizeof(half_t);  // 2 MB
  half_t* xt = (half_t*)(ws);
  half_t* qn = (half_t*)(ws + SZ);
  half_t* kt = (half_t*)(ws + 2 * SZ);
  half_t* vh = (half_t*)(ws + 3 * SZ);
  half_t* aout = (half_t*)(ws + 4 * SZ);
  half_t* wqh = (half_t*)(ws + 5 * SZ);
  half_t* wph = (half_t*)(ws + 5 * SZ + 32768);

  cvt_kernel<<<(BB * CC * NN + 255) / 256, 256, 0, stream>>>(x, wqkv, wproj, xt,
                                                             wqh, wph);
  qkv_kernel<<<(BB * 12 * (NN / 16)) / 8, 256, 0, stream>>>(xt, wqh, bqkv, qn,
                                                            kt, vh);
  attn_kernel<<<(BB * (NN / 16)) / 8, 256, 0, stream>>>(qn, kt, vh, aout);
  proj_kernel<<<(BB * 4 * (NN / 16)) / 8, 256, 0, stream>>>(aout, wph, bproj, x,
                                                            out);
}
